// EfficientDGCNNBackbone_43946105372956
// MI455X (gfx1250) — matrix operand order repaired and checked by reading, not compiled, not run on hardware
//
#include <hip/hip_runtime.h>

#define EPSV 1e-5f
#define NEG_BIG (-3.4e38f)

typedef __attribute__((ext_vector_type(16))) __bf16 v16bf;
typedef __attribute__((ext_vector_type(8)))  float  v8f;

__device__ __forceinline__ v8f wmma_bf16(v16bf a, v16bf b, v8f c) {
  return __builtin_amdgcn_wmma_f32_16x16x32_bf16(false, a, false, b, (short)0, c, false, false);
}

// A-matrix 16x32 bf16 fragment (ISA 7.12.2): lane half h, pairs p:
//   p=0..3 -> K = 8h + 2p + e ; p=4..7 -> K = 16 + 8h + 2(p-4) + e
// => two contiguous 8-float runs at [8h, 8h+8) and [16+8h, 16+8h+8)
__device__ __forceinline__ v16bf load_bf16_A(const float* __restrict__ p, int half) {
  const float4 a0 = *(const float4*)(p + 8 * half);
  const float4 a1 = *(const float4*)(p + 8 * half + 4);
  const float4 b0 = *(const float4*)(p + 16 + 8 * half);
  const float4 b1 = *(const float4*)(p + 16 + 8 * half + 4);
  v16bf v;
  v[0] = (__bf16)a0.x; v[1] = (__bf16)a0.y; v[2] = (__bf16)a0.z; v[3] = (__bf16)a0.w;
  v[4] = (__bf16)a1.x; v[5] = (__bf16)a1.y; v[6] = (__bf16)a1.z; v[7] = (__bf16)a1.w;
  v[8] = (__bf16)b0.x; v[9] = (__bf16)b0.y; v[10] = (__bf16)b0.z; v[11] = (__bf16)b0.w;
  v[12] = (__bf16)b1.x; v[13] = (__bf16)b1.y; v[14] = (__bf16)b1.z; v[15] = (__bf16)b1.w;
  return v;
}

// B-matrix 32x16 bf16 fragment: lane half h covers K = [8h, 8h+8) and [16+8h, 16+8h+8) (same as A)
__device__ __forceinline__ v16bf load_bf16_B(const float* __restrict__ p, int half) {
  const float4 a0 = *(const float4*)(p + 8 * half);
  const float4 a1 = *(const float4*)(p + 8 * half + 4);
  const float4 a2 = *(const float4*)(p + 16 + 8 * half);
  const float4 a3 = *(const float4*)(p + 16 + 8 * half + 4);
  v16bf v;
  v[0] = (__bf16)a0.x; v[1] = (__bf16)a0.y; v[2] = (__bf16)a0.z; v[3] = (__bf16)a0.w;
  v[4] = (__bf16)a1.x; v[5] = (__bf16)a1.y; v[6] = (__bf16)a1.z; v[7] = (__bf16)a1.w;
  v[8] = (__bf16)a2.x; v[9] = (__bf16)a2.y; v[10] = (__bf16)a2.z; v[11] = (__bf16)a2.w;
  v[12] = (__bf16)a3.x; v[13] = (__bf16)a3.y; v[14] = (__bf16)a3.z; v[15] = (__bf16)a3.w;
  return v;
}

// ---------------- pack: (B,3,N) channel-major -> point-major padded to 32ch; sqnorm ----------
__global__ void pack_kernel(const float* __restrict__ xin, float* __restrict__ X0,
                            float* __restrict__ xx, int Nn) {
  int n = blockIdx.x * 256 + threadIdx.x;
  int b = blockIdx.y;
  float v0 = xin[((size_t)b * 3 + 0) * Nn + n];
  float v1 = xin[((size_t)b * 3 + 1) * Nn + n];
  float v2 = xin[((size_t)b * 3 + 2) * Nn + n];
  float* o = X0 + ((size_t)b * Nn + n) * 32;
  o[0] = v0; o[1] = v1; o[2] = v2;
#pragma unroll
  for (int c = 3; c < 32; ++c) o[c] = 0.f;
  xx[(size_t)b * Nn + n] = v0 * v0 + v1 * v1 + v2 * v2;
}

// ---------------- weight prep: W1 = w[:, :C], Wc = w[:, C:] - w[:, :C], zero-padded ----------
__global__ void prepw_kernel(const float* __restrict__ w, int O, int C, int Cpad,
                             float* __restrict__ W1, float* __restrict__ Wc) {
  int t = blockIdx.x * 256 + threadIdx.x;
  if (t >= O * Cpad) return;
  int o = t / Cpad, c = t % Cpad;
  float a = 0.f, d = 0.f;
  if (c < C) {
    a = w[(size_t)o * 2 * C + c];
    d = w[(size_t)o * 2 * C + C + c] - a;
  }
  W1[t] = a; Wc[t] = d;
}

// ---------------- squared norms over a channel slice ----------------
__global__ void sqnorm_kernel(const float* __restrict__ X, int ldx, int C,
                              float* __restrict__ xx, int Nn) {
  int n = blockIdx.x * 256 + threadIdx.x;
  int b = blockIdx.y;
  const float* p = X + ((size_t)b * Nn + n) * ldx;
  float s = 0.f;
  for (int c = 0; c < C; ++c) s += p[c] * p[c];
  xx[(size_t)b * Nn + n] = s;
}

// ---------------- KNN: streaming top-20; candidate tiles staged via async global->LDS -------
template <int C>
__global__ __launch_bounds__(256) void knn_kernel(const float* __restrict__ X, int ldx,
                                                  const float* __restrict__ xx,
                                                  int* __restrict__ out, int Nn) {
  __shared__ __align__(16) float tile[32 * C];
  __shared__ float txx[32];
  const int tid = threadIdx.x;
  const int b = blockIdx.y;
  const int q = blockIdx.x * 256 + tid;
  const float* Xb = X + (size_t)b * Nn * ldx;
  float qf[C];
  {
    const float* qp = Xb + (size_t)q * ldx;
#pragma unroll
    for (int c = 0; c < C; ++c) qf[c] = qp[c];
  }
  float best[20]; int bidx[20];
#pragma unroll
  for (int k = 0; k < 20; ++k) { best[k] = NEG_BIG; bidx[k] = 0; }

  const unsigned lds_base = (unsigned)(size_t)&tile[0];  // generic LDS ptr: low 32b = offset
  const int CPR = C / 4;  // 16B chunks per candidate row

  for (int tb = 0; tb < Nn; tb += 32) {
    // async stage: 32 rows x C floats, 16B per lane-issue (ASYNCcnt-tracked, no VGPR bounce)
    for (int e = tid; e < 8 * C; e += 256) {
      int j = e / CPR, cc = e % CPR;
      const float* src = Xb + (size_t)(tb + j) * ldx + cc * 4;
      unsigned long long ga = (unsigned long long)(size_t)src;
      unsigned ldsoff = lds_base + (unsigned)e * 16u;
      asm volatile("global_load_async_to_lds_b128 %0, %1, off"
                   :: "v"(ldsoff), "v"(ga) : "memory");
    }
    if (tid < 32) txx[tid] = xx[(size_t)b * Nn + tb + tid];
    asm volatile("s_wait_asynccnt 0x0" ::: "memory");
    __syncthreads();
    for (int j = 0; j < 32; ++j) {
      const float4* tp = (const float4*)(tile + j * C);
      float dot = 0.f;
#pragma unroll
      for (int c4 = 0; c4 < C / 4; ++c4) {
        float4 tv = tp[c4];
        dot += qf[4 * c4 + 0] * tv.x + qf[4 * c4 + 1] * tv.y +
               qf[4 * c4 + 2] * tv.z + qf[4 * c4 + 3] * tv.w;
      }
      float s = 2.f * dot - txx[j];
      if (s > best[19]) {
        int p = 19;
        while (p > 0 && best[p - 1] < s) {
          best[p] = best[p - 1]; bidx[p] = bidx[p - 1]; --p;
        }
        best[p] = s; bidx[p] = tb + j;
      }
    }
    __syncthreads();
  }
  int* op = out + ((size_t)b * Nn + q) * 20;
#pragma unroll
  for (int k = 0; k < 20; ++k) op[k] = bidx[k];
}

// ---------------- EdgeConv: Y[o,n] = max_k leaky(bn( W1@x[idx] + (W2-W1)@x[n] )) ------------
// one wave per (16 out-ch, 16 points); center term = WMMA accumulator seed
template <int CT>
__global__ __launch_bounds__(32) void edgeconv_kernel(
    const float* __restrict__ X, int ldx, const int* __restrict__ nidx,
    const float* __restrict__ W1, const float* __restrict__ Wc,
    const float* __restrict__ bng, const float* __restrict__ bnb,
    const float* __restrict__ bnm, const float* __restrict__ bnv,
    float* __restrict__ Y, int ldy, int Nn) {
  const int C = CT * 32;
  const int lane = threadIdx.x & 31;
  const int half = lane >> 4;
  const int l16 = lane & 15;
  const int o0 = blockIdx.x * 16;
  const int n0 = blockIdx.y * 16;
  const int b = blockIdx.z;
  const float* Xb = X + (size_t)b * Nn * ldx;

  v16bf a1[CT], ac[CT];
  {
    const float* r1 = W1 + (size_t)(o0 + l16) * C;
    const float* rc = Wc + (size_t)(o0 + l16) * C;
#pragma unroll
    for (int t = 0; t < CT; ++t) {
      a1[t] = load_bf16_A(r1 + t * 32, half);
      ac[t] = load_bf16_A(rc + t * 32, half);
    }
  }
  v8f tacc = {0.f, 0.f, 0.f, 0.f, 0.f, 0.f, 0.f, 0.f};
  {
    const float* xc = Xb + (size_t)(n0 + l16) * ldx;
#pragma unroll
    for (int t = 0; t < CT; ++t) tacc = wmma_bf16(ac[t], load_bf16_B(xc + t * 32, half), tacc);
  }
  float sc[8], bi[8];
  {
    const int ob = o0 + 8 * half;
#pragma unroll
    for (int r = 0; r < 8; ++r) {
      float g = bng[ob + r], be = bnb[ob + r], m = bnm[ob + r], vv = bnv[ob + r];
      float s = g * rsqrtf(vv + EPSV);
      sc[r] = s; bi[r] = be - m * s;
    }
  }
  const int* ip = nidx + ((size_t)b * Nn + n0 + l16) * 20;
  float best[8];
#pragma unroll
  for (int r = 0; r < 8; ++r) best[r] = NEG_BIG;

  for (int k = 0; k < 20; ++k) {
    int m = ip[k];
    if (k < 19) __builtin_prefetch(Xb + (size_t)ip[k + 1] * ldx, 0, 3);
    const float* xn = Xb + (size_t)m * ldx;
    v8f acc = tacc;
#pragma unroll
    for (int t = 0; t < CT; ++t) acc = wmma_bf16(a1[t], load_bf16_B(xn + t * 32, half), acc);
#pragma unroll
    for (int r = 0; r < 8; ++r) {
      float h = acc[r] * sc[r] + bi[r];
      h = fmaxf(h, 0.2f * h);
      best[r] = fmaxf(best[r], h);
    }
  }
  float* yp = Y + ((size_t)b * Nn + n0 + l16) * ldy + o0 + 8 * half;
  float4 s0 = make_float4(best[0], best[1], best[2], best[3]);
  float4 s1 = make_float4(best[4], best[5], best[6], best[7]);
  *(float4*)yp = s0;
  *(float4*)(yp + 4) = s1;
}

// ---------------- generic WMMA GEMM: 16x64 output per wave (4 col tiles share each A tile) --
template <int CT>
__global__ __launch_bounds__(32) void gemm_kernel(
    const float* __restrict__ A, int lda, int Oreal,
    const float* __restrict__ Bsrc, int ldb,
    const float* __restrict__ extra, int splitC,
    const float* __restrict__ bias,
    const float* __restrict__ bng, const float* __restrict__ bnb,
    const float* __restrict__ bnm, const float* __restrict__ bnv,
    int leaky, float* __restrict__ Y, int ldy, int Nn) {
  const int lane = threadIdx.x & 31;
  const int half = lane >> 4;
  const int l16 = lane & 15;
  const int o0 = blockIdx.x * 16;
  const int n0 = blockIdx.y * 64;
  const int b = blockIdx.z;
  int arow = o0 + l16;
  if (arow >= Oreal) arow = Oreal - 1;  // padded rows: load valid row, never stored
  const float* Ar = A + (size_t)arow * lda;
  const float* xc[4];
#pragma unroll
  for (int j = 0; j < 4; ++j)
    xc[j] = Bsrc + ((size_t)b * Nn + n0 + 16 * j + l16) * ldb;
  const float* ex = extra ? (extra + (size_t)b * 512) : nullptr;

  v8f z = {0.f, 0.f, 0.f, 0.f, 0.f, 0.f, 0.f, 0.f};
  v8f acc[4] = {z, z, z, z};
  for (int t = 0; t < CT; ++t) {
    const int cbase = t * 32;
    v16bf at = load_bf16_A(Ar + cbase, half);
    if (cbase >= splitC) {
      // broadcast-g virtual channels: B fragment column-invariant -> one load, 4 WMMAs
      v16bf bt = load_bf16_B(ex + (cbase - splitC), half);
#pragma unroll
      for (int j = 0; j < 4; ++j) acc[j] = wmma_bf16(at, bt, acc[j]);
    } else {
#pragma unroll
      for (int j = 0; j < 4; ++j) acc[j] = wmma_bf16(at, load_bf16_B(xc[j] + cbase, half), acc[j]);
    }
  }
  const int ob = o0 + 8 * half;
  float addv[8], scv[8], biv[8];
  bool okr[8];
#pragma unroll
  for (int r = 0; r < 8; ++r) {
    int o = ob + r;
    bool ok = o < Oreal;
    okr[r] = ok;
    addv[r] = (bias && ok) ? bias[o] : 0.f;
    if (bng && ok) {
      float s = bng[o] * rsqrtf(bnv[o] + EPSV);
      scv[r] = s; biv[r] = bnb[o] - bnm[o] * s;
    } else {
      scv[r] = 1.f; biv[r] = 0.f;
    }
  }
#pragma unroll
  for (int j = 0; j < 4; ++j) {
    float* yp = Y + ((size_t)b * Nn + n0 + 16 * j + l16) * ldy;
#pragma unroll
    for (int r = 0; r < 8; ++r) {
      float v = (acc[j][r] + addv[r]) * scv[r] + biv[r];
      if (leaky) v = fmaxf(v, 0.2f * v);
      if (okr[r]) yp[ob + r] = v;
    }
  }
}

// ---------------- global max over points per channel (two-stage for parallelism) ------------
__global__ __launch_bounds__(256) void gmax1_kernel(const float* __restrict__ H,
                                                    float* __restrict__ gpart, int Nn) {
  int o = blockIdx.x * 256 + threadIdx.x;
  int chunk = blockIdx.y;
  int b = blockIdx.z;
  const int nch = Nn / 16;
  const float* p = H + (size_t)b * Nn * 512 + (size_t)chunk * nch * 512 + o;
  float m = NEG_BIG;
  for (int n = 0; n < nch; ++n) m = fmaxf(m, p[(size_t)n * 512]);
  gpart[((size_t)b * 16 + chunk) * 512 + o] = m;
}
__global__ __launch_bounds__(256) void gmax2_kernel(const float* __restrict__ gpart,
                                                    float* __restrict__ g) {
  int o = blockIdx.x * 256 + threadIdx.x;
  int b = blockIdx.y;
  float m = NEG_BIG;
#pragma unroll
  for (int c = 0; c < 16; ++c) m = fmaxf(m, gpart[((size_t)b * 16 + c) * 512 + o]);
  g[b * 512 + o] = m;
}

extern "C" void kernel_launch(void* const* d_in, const int* in_sizes, int n_in,
                              void* d_out, int out_size, void* d_ws, size_t ws_size,
                              hipStream_t stream) {
  const int B = 2, Nn = 8192;
  (void)in_sizes; (void)n_in; (void)out_size;

  const float* xin = (const float*)d_in[0];
  const float* c1w = (const float*)d_in[1];
  const float* bn1g = (const float*)d_in[2], *bn1b = (const float*)d_in[3],
             *bn1m = (const float*)d_in[4], *bn1v = (const float*)d_in[5];
  const float* c2w = (const float*)d_in[6];
  const float* bn2g = (const float*)d_in[7], *bn2b = (const float*)d_in[8],
             *bn2m = (const float*)d_in[9], *bn2v = (const float*)d_in[10];
  const float* c3w = (const float*)d_in[11];
  const float* bn3g = (const float*)d_in[12], *bn3b = (const float*)d_in[13],
             *bn3m = (const float*)d_in[14], *bn3v = (const float*)d_in[15];
  const float* c4w = (const float*)d_in[16];
  const float* bn4g = (const float*)d_in[17], *bn4b = (const float*)d_in[18],
             *bn4m = (const float*)d_in[19], *bn4v = (const float*)d_in[20];
  const float* c5w = (const float*)d_in[21];
  const float* bn5g = (const float*)d_in[22], *bn5b = (const float*)d_in[23],
             *bn5m = (const float*)d_in[24], *bn5v = (const float*)d_in[25];
  const float* fw[4] = {(const float*)d_in[26], (const float*)d_in[27],
                        (const float*)d_in[28], (const float*)d_in[29]};
  const float* fb[4] = {(const float*)d_in[30], (const float*)d_in[31],
                        (const float*)d_in[32], (const float*)d_in[33]};
  const float* fbn[4][4];
  for (int i = 0; i < 4; ++i)
    for (int j = 0; j < 4; ++j) fbn[i][j] = (const float*)d_in[34 + i * 4 + j];
  const float* semw = (const float*)d_in[50];
  const float* semb = (const float*)d_in[51];

  // workspace layout (floats)
  size_t need = (size_t)B * Nn * 32 + (size_t)B * Nn * 512 * 2 + B * 512 + (size_t)B * 16 * 512 +
                (size_t)B * Nn + (size_t)B * Nn * 20 +
                (64 * 32 + 64 * 64 + 128 * 64 + 256 * 128) * 2;
  if (ws_size < need * 4) return;
  float* X0 = (float*)d_ws;
  float* XC = X0 + (size_t)B * Nn * 32;
  float* H = XC + (size_t)B * Nn * 512;
  float* gv = H + (size_t)B * Nn * 512;
  float* gpart = gv + B * 512;
  float* xx = gpart + (size_t)B * 16 * 512;
  int* idx = (int*)(xx + (size_t)B * Nn);
  float* wp = (float*)(idx + (size_t)B * Nn * 20);
  float* w11 = wp;               float* wc1 = w11 + 64 * 32;
  float* w12 = wc1 + 64 * 32;    float* wc2 = w12 + 64 * 64;
  float* w13 = wc2 + 64 * 64;    float* wc3 = w13 + 128 * 64;
  float* w14 = wc3 + 128 * 64;   float* wc4 = w14 + 256 * 128;

  pack_kernel<<<dim3(Nn / 256, B), 256, 0, stream>>>(xin, X0, xx, Nn);
  prepw_kernel<<<dim3((64 * 32 + 255) / 256), 256, 0, stream>>>(c1w, 64, 3, 32, w11, wc1);
  prepw_kernel<<<dim3((64 * 64 + 255) / 256), 256, 0, stream>>>(c2w, 64, 64, 64, w12, wc2);
  prepw_kernel<<<dim3((128 * 64 + 255) / 256), 256, 0, stream>>>(c3w, 128, 64, 64, w13, wc3);
  prepw_kernel<<<dim3((256 * 128 + 255) / 256), 256, 0, stream>>>(c4w, 256, 128, 128, w14, wc4);

  // Layer 1: C=3 (padded 32) -> 64
  knn_kernel<32><<<dim3(Nn / 256, B), 256, 0, stream>>>(X0, 32, xx, idx, Nn);
  edgeconv_kernel<1><<<dim3(4, Nn / 16, B), 32, 0, stream>>>(
      X0, 32, idx, w11, wc1, bn1g, bn1b, bn1m, bn1v, XC + 0, 512, Nn);
  // Layer 2: 64 -> 64
  sqnorm_kernel<<<dim3(Nn / 256, B), 256, 0, stream>>>(XC + 0, 512, 64, xx, Nn);
  knn_kernel<64><<<dim3(Nn / 256, B), 256, 0, stream>>>(XC + 0, 512, xx, idx, Nn);
  edgeconv_kernel<2><<<dim3(4, Nn / 16, B), 32, 0, stream>>>(
      XC + 0, 512, idx, w12, wc2, bn2g, bn2b, bn2m, bn2v, XC + 64, 512, Nn);
  // Layer 3: 64 -> 128
  sqnorm_kernel<<<dim3(Nn / 256, B), 256, 0, stream>>>(XC + 64, 512, 64, xx, Nn);
  knn_kernel<64><<<dim3(Nn / 256, B), 256, 0, stream>>>(XC + 64, 512, xx, idx, Nn);
  edgeconv_kernel<2><<<dim3(8, Nn / 16, B), 32, 0, stream>>>(
      XC + 64, 512, idx, w13, wc3, bn3g, bn3b, bn3m, bn3v, XC + 128, 512, Nn);
  // Layer 4: 128 -> 256
  sqnorm_kernel<<<dim3(Nn / 256, B), 256, 0, stream>>>(XC + 128, 512, 128, xx, Nn);
  knn_kernel<128><<<dim3(Nn / 256, B), 256, 0, stream>>>(XC + 128, 512, xx, idx, Nn);
  edgeconv_kernel<4><<<dim3(16, Nn / 16, B), 32, 0, stream>>>(
      XC + 128, 512, idx, w14, wc4, bn4g, bn4b, bn4m, bn4v, XC + 256, 512, Nn);

  // conv5: 512 -> 512 + bn + leaky (16x64 per wave)
  gemm_kernel<16><<<dim3(32, Nn / 64, B), 32, 0, stream>>>(
      c5w, 512, 512, XC, 512, nullptr, 1 << 30, nullptr,
      bn5g, bn5b, bn5m, bn5v, 1, H, 512, Nn);
  gmax1_kernel<<<dim3(2, 16, B), 256, 0, stream>>>(H, gpart, Nn);
  gmax2_kernel<<<dim3(2, B), 256, 0, stream>>>(gpart, gv);

  // feature heads: hh = [h ; broadcast g] (1024 ch), + bias, bn; write into 692-concat
  float* out = (float*)d_out;
  const int offs[4] = {0, 256, 448, 576};
  const int oc[4] = {256, 192, 128, 96};
  for (int i = 0; i < 4; ++i) {
    gemm_kernel<32><<<dim3((oc[i] + 15) / 16, Nn / 64, B), 32, 0, stream>>>(
        fw[i], 1024, oc[i], H, 512, gv, 512, fb[i],
        fbn[i][0], fbn[i][1], fbn[i][2], fbn[i][3], 0, out + offs[i], 692, Nn);
  }
  // sem: feats[-1] (96ch at offset 576) @ sem_w^T + sem_b -> 20ch at offset 672
  gemm_kernel<3><<<dim3(2, Nn / 64, B), 32, 0, stream>>>(
      semw, 96, 20, out + 576, 692, nullptr, 1 << 30, semb,
      nullptr, nullptr, nullptr, nullptr, 0, out + 672, 692, Nn);
}
